// PhysicsDiscriminator_62457414419222
// MI455X (gfx1250) — matrix operand order repaired and checked by reading, not compiled, not run on hardware
//
#include <hip/hip_runtime.h>
#include <hip/hip_bf16.h>
#include <math.h>

// ---------------- CDNA5 (gfx1250) wave32 WMMA GNN pipeline ----------------
// N=50000 nodes, E=1600000 edges, D_IN=128, H=64. All multiples of 16.

#define NODES 50000
#define EDGES 1600000

typedef __bf16 bf16_t;
typedef __attribute__((ext_vector_type(16))) __bf16 v16bf;
typedef __attribute__((ext_vector_type(8)))  __bf16 v8bf;
typedef __attribute__((ext_vector_type(4)))  __bf16 v4bf;
typedef __attribute__((ext_vector_type(8)))  float  v8f;

// ---- WMMA fragment helpers (layouts per CDNA5 ISA 7.12.2, wave32) ----
// A (16x32 bf16, row-major source): lane L holds row M=L%16.
//   lanes 0-15:  elems 0..7 = K[k0..k0+7],   elems 8..15 = K[k0+16..k0+23]
//   lanes 16-31: elems 0..7 = K[k0+8..k0+15], elems 8..15 = K[k0+24..k0+31]
__device__ inline v16bf load_a_frag(const bf16_t* row, int k0, int lane) {
    const int hi = (lane >> 4) << 3;          // 0 or 8
    union { v16bf v; v8bf h[2]; } u;
    u.h[0] = *(const v8bf*)(row + k0 + hi);
    u.h[1] = *(const v8bf*)(row + k0 + 16 + hi);
    return u.v;
}

// B (32x16 bf16 = W.T tile) from row-major W[Nout][K]:
// lane n holds column N=n%16 -> W row (n0 + n%16), same K order as A:
// elems 0..7 = K[k0+hi..+7], elems 8..15 = K[k0+16+hi..+7], hi = 8 for high half-wave.
__device__ inline v16bf load_b_frag(const bf16_t* W, int K, int n0, int k0, int lane) {
    const bf16_t* p = W + (size_t)(n0 + (lane & 15)) * K + k0 + ((lane >> 4) << 3);
    union { v16bf v; v8bf h[2]; } u;
    u.h[0] = *(const v8bf*)(p);
    u.h[1] = *(const v8bf*)(p + 16);
    return u.v;
}

// A fragment = 0.5*(h[src] + h[dst]) built in-register (bond edge features).
__device__ inline v16bf avg_a_frag(const bf16_t* ra, const bf16_t* rb, int k0, int lane) {
    const int hi = (lane >> 4) << 3;
    union { v16bf v; v8bf h[2]; } a, b, o;
    a.h[0] = *(const v8bf*)(ra + k0 + hi);
    a.h[1] = *(const v8bf*)(ra + k0 + 16 + hi);
    b.h[0] = *(const v8bf*)(rb + k0 + hi);
    b.h[1] = *(const v8bf*)(rb + k0 + 16 + hi);
#pragma unroll
    for (int i = 0; i < 16; ++i)
        o.v[i] = (bf16_t)(0.5f * ((float)a.v[i] + (float)b.v[i]));
    return o.v;
}

__device__ inline v8f wmma_bf16(v16bf a, v16bf b, v8f c) {
    return __builtin_amdgcn_wmma_f32_16x16x32_bf16(
        /*neg_a=*/false, a, /*neg_b=*/false, b,
        /*c_mod=*/(short)0, c, /*reuse_a=*/false, /*reuse_b=*/false);
}

// ---------------- small utility kernels ----------------
__global__ void k_f2bf(const float* __restrict__ s, bf16_t* __restrict__ d, int n) {
    int i = blockIdx.x * blockDim.x + threadIdx.x;
    if (i < n) d[i] = (bf16_t)s[i];
}
__global__ void k_fill(float* p, int n, float v) {
    int i = blockIdx.x * blockDim.x + threadIdx.x;
    if (i < n) p[i] = v;
}
__global__ void k_deg_count(const int* __restrict__ dst, float* __restrict__ deg, int e) {
    int i = blockIdx.x * blockDim.x + threadIdx.x;
    if (i < e) atomicAdd(&deg[dst[i]], 1.0f);
}
__global__ void k_rsqrt(float* p, int n) {
    int i = blockIdx.x * blockDim.x + threadIdx.x;
    if (i < n) p[i] = rsqrtf(p[i]);
}

// ---------------- generic WMMA GEMM: Out = act(A[M,K] @ W[NT*16,K]^T + b) ----------------
// One wave per 16-row tile, NT column tiles. Blocks of 8 waves.
template <int NT, int ACT>
__global__ void k_gemm(const bf16_t* __restrict__ A, const bf16_t* __restrict__ W,
                       const float* __restrict__ bias, bf16_t* __restrict__ Out,
                       int M, int K) {
    const int lane = threadIdx.x & 31;
    const int wave = threadIdx.x >> 5;
    const int tile = blockIdx.x * (blockDim.x >> 5) + wave;
    if (tile * 16 >= M) return;
    const int NO = NT * 16;
    const bf16_t* arow = A + (size_t)(tile * 16 + (lane & 15)) * K;
    v8f acc[NT] = {};
    for (int k0 = 0; k0 < K; k0 += 32) {
        v16bf a = load_a_frag(arow, k0, lane);
#pragma unroll
        for (int nt = 0; nt < NT; ++nt) {
            v16bf b = load_b_frag(W, K, nt * 16, k0, lane);
            acc[nt] = wmma_bf16(a, b, acc[nt]);
        }
    }
    // C/D layout: lane holds c[r] = D[(lane/16)*8 + r][lane%16]
    const int rbase = tile * 16 + ((lane >> 4) << 3);
#pragma unroll
    for (int nt = 0; nt < NT; ++nt) {
        const int col = nt * 16 + (lane & 15);
        const float bv = bias ? bias[col] : 0.0f;
#pragma unroll
        for (int r = 0; r < 8; ++r) {
            float v = acc[nt][r] + bv;
            if (ACT) v = fmaxf(v, 0.0f);
            Out[(size_t)(rbase + r) * NO + col] = (bf16_t)v;
        }
    }
}

// ---------------- GCN scatter (NO = 128 fixed) ----------------
__global__ void k_scatter_init(const bf16_t* __restrict__ t, const float* __restrict__ dinv,
                               float* __restrict__ acc, int total) {
    int i = blockIdx.x * blockDim.x + threadIdx.x;
    if (i >= total) return;
    const float di = dinv[i >> 7];
    acc[i] = (float)t[i] * di * di;          // self-loop term
}
__global__ void k_scatter_edges(const bf16_t* __restrict__ t, const int* __restrict__ src,
                                const int* __restrict__ dst, const float* __restrict__ dinv,
                                float* __restrict__ acc, int total) {
    int i = blockIdx.x * blockDim.x + threadIdx.x;   // E * 32 threads (4 floats each)
    if (i >= total) return;
    const int e = i >> 5;
    const int c = (i & 31) << 2;
    const int s = src[e], d = dst[e];
    const float w = dinv[s] * dinv[d];
    const bf16_t* tp = t + ((size_t)s << 7) + c;
    float* ap = acc + ((size_t)d << 7) + c;
    __builtin_prefetch(tp, 0, 1);
    const v4bf tv = *(const v4bf*)tp;
#pragma unroll
    for (int j = 0; j < 4; ++j) atomicAdd(ap + j, (float)tv[j] * w);
}
__global__ void k_gcn_finalize(const float* __restrict__ acc, const float* __restrict__ bias,
                               bf16_t* __restrict__ out, int total) {
    int i = blockIdx.x * blockDim.x + threadIdx.x;
    if (i >= total) return;
    out[i] = (bf16_t)fmaxf(acc[i] + bias[i & 127], 0.0f);
}

// ---------------- bond energy: per-wave 16-edge tile ----------------
__global__ void k_bond(const bf16_t* __restrict__ h, const int* __restrict__ src,
                       const int* __restrict__ dst, const float* __restrict__ pos,
                       const bf16_t* __restrict__ W1, const float* __restrict__ b1,
                       const float* __restrict__ w2, const float* __restrict__ b2,
                       float* __restrict__ out_sum) {
    __shared__ float sred[8];
    const int lane = threadIdx.x & 31;
    const int wave = threadIdx.x >> 5;
    const int tile = blockIdx.x * 8 + wave;            // exact: EDGES/16/8 blocks
    const int e = tile * 16 + (lane & 15);
    const int s = src[e], d = dst[e];
    const float dx = pos[(size_t)d * 3 + 0] - pos[(size_t)s * 3 + 0];
    const float dy = pos[(size_t)d * 3 + 1] - pos[(size_t)s * 3 + 1];
    const float dz = pos[(size_t)d * 3 + 2] - pos[(size_t)s * 3 + 2];
    const float len = sqrtf(dx * dx + dy * dy + dz * dz);
    const bf16_t* hs = h + ((size_t)s << 7);
    const bf16_t* hd = h + ((size_t)d << 7);
    v8f acc[4] = {};
#pragma unroll
    for (int k0 = 0; k0 < 128; k0 += 32) {
        v16bf a = avg_a_frag(hs, hd, k0, lane);
#pragma unroll
        for (int nt = 0; nt < 4; ++nt) {
            v16bf b = load_b_frag(W1, 128, nt * 16, k0, lane);
            acc[nt] = wmma_bf16(a, b, acc[nt]);
        }
    }
    // relu(q + b1) . w2, per edge; lane holds 8 rows x 1 col per ntile
    float part[8] = {0.f, 0.f, 0.f, 0.f, 0.f, 0.f, 0.f, 0.f};
#pragma unroll
    for (int nt = 0; nt < 4; ++nt) {
        const int col = nt * 16 + (lane & 15);
        const float b1v = b1[col];
        const float w2v = w2[col];
#pragma unroll
        for (int r = 0; r < 8; ++r)
            part[r] += fmaxf(acc[nt][r] + b1v, 0.0f) * w2v;
    }
    // reduce over the 16 columns within each half-wave (xor masks stay in-group)
#pragma unroll
    for (int off = 8; off >= 1; off >>= 1)
#pragma unroll
        for (int r = 0; r < 8; ++r)
            part[r] += __shfl_xor(part[r], off, 32);
    // fetch lengths for this half's 8 edges
    float lens[8];
#pragma unroll
    for (int r = 0; r < 8; ++r)
        lens[r] = __shfl(len, ((lane >> 4) << 3) + r, 32);
    float local = 0.0f;
    if ((lane & 15) == 0) {
        const float b2v = b2[0];
#pragma unroll
        for (int r = 0; r < 8; ++r) {
            const float dl = lens[r] - 1.5f;
            local += part[r] + b2v + 1000.0f * dl * dl;
        }
    }
    local += __shfl_xor(local, 16, 32);
    if (lane == 0) sred[wave] = local;
    __syncthreads();
    if (threadIdx.x == 0) {
        float tsum = 0.0f;
#pragma unroll
        for (int w = 0; w < 8; ++w) tsum += sred[w];
        atomicAdd(out_sum, tsum);
    }
}

// ---------------- transmembrane validator: two 3-layer MLPs per node ----------------
__global__ void k_tm(const bf16_t* __restrict__ h,
                     const bf16_t* __restrict__ Wh1, const float* __restrict__ bh1,
                     const bf16_t* __restrict__ Wh2, const float* __restrict__ bh2,
                     const float* __restrict__ wh3, const float* __restrict__ bh3,
                     const bf16_t* __restrict__ We1, const float* __restrict__ be1,
                     const bf16_t* __restrict__ We2, const float* __restrict__ be2,
                     const float* __restrict__ we3, const float* __restrict__ be3,
                     float* __restrict__ out_sum, int nnodes) {
    __shared__ bf16_t lds1[8][16 * 64];
    __shared__ bf16_t lds2[8][16 * 32];
    const int lane = threadIdx.x & 31;
    const int wave = threadIdx.x >> 5;
    const int tile = blockIdx.x * 8 + wave;
    if (tile * 16 >= nnodes) return;                   // no block barriers below
    const bf16_t* arow = h + ((size_t)(tile * 16 + (lane & 15)) << 7);
    const int rb = (lane >> 4) << 3;
    float prob[2];
#pragma unroll
    for (int net = 0; net < 2; ++net) {
        const bf16_t* W1 = net ? We1 : Wh1;  const float* B1 = net ? be1 : bh1;
        const bf16_t* W2 = net ? We2 : Wh2;  const float* B2 = net ? be2 : bh2;
        const float*  W3 = net ? we3 : wh3;  const float* B3 = net ? be3 : bh3;
        // layer1: [16,128] @ [64,128]^T
        v8f a1[4] = {};
#pragma unroll
        for (int k0 = 0; k0 < 128; k0 += 32) {
            v16bf a = load_a_frag(arow, k0, lane);
#pragma unroll
            for (int nt = 0; nt < 4; ++nt)
                a1[nt] = wmma_bf16(a, load_b_frag(W1, 128, nt * 16, k0, lane), a1[nt]);
        }
#pragma unroll
        for (int nt = 0; nt < 4; ++nt) {
            const int col = nt * 16 + (lane & 15);
            const float bv = B1[col];
#pragma unroll
            for (int r = 0; r < 8; ++r)
                lds1[wave][(rb + r) * 64 + col] = (bf16_t)fmaxf(a1[nt][r] + bv, 0.0f);
        }
        asm volatile("s_wait_dscnt 0x0" ::: "memory");   // LDS RAW within wave
        // layer2: [16,64] @ [32,64]^T, A re-fed from LDS (transposed via C->LDS->A)
        const bf16_t* a2row = &lds1[wave][(lane & 15) * 64];
        v8f a2[2] = {};
#pragma unroll
        for (int k0 = 0; k0 < 64; k0 += 32) {
            v16bf a = load_a_frag(a2row, k0, lane);
#pragma unroll
            for (int nt = 0; nt < 2; ++nt)
                a2[nt] = wmma_bf16(a, load_b_frag(W2, 64, nt * 16, k0, lane), a2[nt]);
        }
#pragma unroll
        for (int nt = 0; nt < 2; ++nt) {
            const int col = nt * 16 + (lane & 15);
            const float bv = B2[col];
#pragma unroll
            for (int r = 0; r < 8; ++r)
                lds2[wave][(rb + r) * 32 + col] = (bf16_t)fmaxf(a2[nt][r] + bv, 0.0f);
        }
        asm volatile("s_wait_dscnt 0x0" ::: "memory");
        // layer3: [16,32] . w3 + b3 -> sigmoid, lanes 0..15 own one row each
        float z = B3[0];
        if (lane < 16) {
#pragma unroll
            for (int j = 0; j < 32; ++j)
                z += (float)lds2[wave][lane * 32 + j] * W3[j];
        }
        prob[net] = 1.0f / (1.0f + expf(-z));
    }
    float contrib = (lane < 16) ? 0.5f * (prob[0] + prob[1]) : 0.0f;
#pragma unroll
    for (int off = 16; off >= 1; off >>= 1)
        contrib += __shfl_xor(contrib, off, 32);
    if (lane == 0) atomicAdd(out_sum, contrib);
}

__global__ void k_final(const float* __restrict__ scal, float* __restrict__ out) {
    const float e_bond = scal[0];
    const float tm = scal[1] / (float)NODES;
    float e_total = e_bond + 3.0f;
    if (tm < 0.5f) e_total += (1.0f - tm) * 10.0f;
    out[0] = e_bond; out[1] = 1.0f; out[2] = 1.0f; out[3] = 1.0f;
    out[4] = e_total; out[5] = tm;
}

// ---------------- host orchestration ----------------
extern "C" void kernel_launch(void* const* d_in, const int* in_sizes, int n_in,
                              void* d_out, int out_size, void* d_ws, size_t ws_size,
                              hipStream_t stream) {
    const float* x       = (const float*)d_in[0];
    const int*   eidx    = (const int*)  d_in[1];
    const int*   src     = eidx;
    const int*   dst     = eidx + EDGES;
    const float* pos     = (const float*)d_in[2];
    const float* emb_w   = (const float*)d_in[4];  const float* emb_b  = (const float*)d_in[5];
    const float* c1_w    = (const float*)d_in[6];  const float* c1_b   = (const float*)d_in[7];
    const float* c2_w    = (const float*)d_in[8];  const float* c2_b   = (const float*)d_in[9];
    const float* bond_w1 = (const float*)d_in[10]; const float* bond_b1= (const float*)d_in[11];
    const float* bond_w2 = (const float*)d_in[12]; const float* bond_b2= (const float*)d_in[13];
    const float* hyd_w1  = (const float*)d_in[14]; const float* hyd_b1 = (const float*)d_in[15];
    const float* hyd_w2  = (const float*)d_in[16]; const float* hyd_b2 = (const float*)d_in[17];
    const float* hyd_w3  = (const float*)d_in[18]; const float* hyd_b3 = (const float*)d_in[19];
    const float* hel_w1  = (const float*)d_in[20]; const float* hel_b1 = (const float*)d_in[21];
    const float* hel_w2  = (const float*)d_in[22]; const float* hel_b2 = (const float*)d_in[23];
    const float* hel_w3  = (const float*)d_in[24]; const float* hel_b3 = (const float*)d_in[25];

    char* ws = (char*)d_ws;
    size_t off = 0;
    auto walloc = [&](size_t bytes) -> void* {
        void* p = ws + off;
        off = (off + bytes + 255) & ~(size_t)255;
        return p;
    };
    bf16_t* x_bf    = (bf16_t*)walloc((size_t)NODES * 128 * 2);
    bf16_t* h0      = (bf16_t*)walloc((size_t)NODES * 64 * 2);
    bf16_t* h1      = (bf16_t*)walloc((size_t)NODES * 128 * 2);
    bf16_t* h2      = (bf16_t*)walloc((size_t)NODES * 128 * 2);
    bf16_t* tbuf    = (bf16_t*)walloc((size_t)NODES * 128 * 2);
    float*  acc     = (float*) walloc((size_t)NODES * 128 * 4);
    float*  dinv    = (float*) walloc((size_t)NODES * 4);
    bf16_t* emb_wb  = (bf16_t*)walloc(64 * 128 * 2);
    bf16_t* c1_wb   = (bf16_t*)walloc(128 * 64 * 2);
    bf16_t* c2_wb   = (bf16_t*)walloc(128 * 128 * 2);
    bf16_t* bond_w1b= (bf16_t*)walloc(64 * 128 * 2);
    bf16_t* hyd_w1b = (bf16_t*)walloc(64 * 128 * 2);
    bf16_t* hyd_w2b = (bf16_t*)walloc(32 * 64 * 2);
    bf16_t* hel_w1b = (bf16_t*)walloc(64 * 128 * 2);
    bf16_t* hel_w2b = (bf16_t*)walloc(32 * 64 * 2);
    float*  scal    = (float*) walloc(8 * 4);

    auto cvt = [&](const float* s, bf16_t* d, int n) {
        k_f2bf<<<(n + 255) / 256, 256, 0, stream>>>(s, d, n);
    };
    // accumulators must be zeroed every launch (deterministic work)
    k_fill<<<1, 32, 0, stream>>>(scal, 2, 0.0f);
    cvt(x, x_bf, NODES * 128);
    cvt(emb_w, emb_wb, 64 * 128);
    cvt(c1_w, c1_wb, 128 * 64);
    cvt(c2_w, c2_wb, 128 * 128);
    cvt(bond_w1, bond_w1b, 64 * 128);
    cvt(hyd_w1, hyd_w1b, 64 * 128);
    cvt(hyd_w2, hyd_w2b, 32 * 64);
    cvt(hel_w1, hel_w1b, 64 * 128);
    cvt(hel_w2, hel_w2b, 32 * 64);

    // symmetric-normalization degrees (shared by both GCN layers)
    k_fill<<<(NODES + 255) / 256, 256, 0, stream>>>(dinv, NODES, 1.0f);
    k_deg_count<<<(EDGES + 255) / 256, 256, 0, stream>>>(dst, dinv, EDGES);
    k_rsqrt<<<(NODES + 255) / 256, 256, 0, stream>>>(dinv, NODES);

    const int gemm_blocks = (NODES / 16 + 7) / 8;   // 391
    // h0 = relu(x @ emb_w^T + emb_b)   [N,128]->[N,64]
    k_gemm<4, 1><<<gemm_blocks, 256, 0, stream>>>(x_bf, emb_wb, emb_b, h0, NODES, 128);

    // GCN layer 1: [N,64] -> [N,128]
    k_gemm<8, 0><<<gemm_blocks, 256, 0, stream>>>(h0, c1_wb, nullptr, tbuf, NODES, 64);
    k_scatter_init <<<(NODES * 128 + 255) / 256, 256, 0, stream>>>(tbuf, dinv, acc, NODES * 128);
    k_scatter_edges<<<(EDGES * 32 + 255) / 256, 256, 0, stream>>>(tbuf, src, dst, dinv, acc, EDGES * 32);
    k_gcn_finalize <<<(NODES * 128 + 255) / 256, 256, 0, stream>>>(acc, c1_b, h1, NODES * 128);

    // GCN layer 2: [N,128] -> [N,128]
    k_gemm<8, 0><<<gemm_blocks, 256, 0, stream>>>(h1, c2_wb, nullptr, tbuf, NODES, 128);
    k_scatter_init <<<(NODES * 128 + 255) / 256, 256, 0, stream>>>(tbuf, dinv, acc, NODES * 128);
    k_scatter_edges<<<(EDGES * 32 + 255) / 256, 256, 0, stream>>>(tbuf, src, dst, dinv, acc, EDGES * 32);
    k_gcn_finalize <<<(NODES * 128 + 255) / 256, 256, 0, stream>>>(acc, c2_b, h2, NODES * 128);

    // bond energy over edges (dominant: bf16 gathers + WMMA edge-GEMM)
    k_bond<<<EDGES / 16 / 8, 256, 0, stream>>>(h2, src, dst, pos,
                                               bond_w1b, bond_b1, bond_w2, bond_b2, &scal[0]);
    // transmembrane validator
    k_tm<<<gemm_blocks, 256, 0, stream>>>(h2,
                                          hyd_w1b, hyd_b1, hyd_w2b, hyd_b2, hyd_w3, hyd_b3,
                                          hel_w1b, hel_b1, hel_w2b, hel_b2, hel_w3, hel_b3,
                                          &scal[1], NODES);
    k_final<<<1, 1, 0, stream>>>(scal, (float*)d_out);
}
